// rt_nw_86380382257715
// MI455X (gfx1250) — matrix operand order repaired and checked by reading, not compiled, not run on hardware
//
#include <hip/hip_runtime.h>
#include <hip/hip_bf16.h>
#include <stdint.h>

#define B_SZ   4096
#define E_SZ   128
#define IP_SZ  64
#define H_SZ   100
#define HP     112          // H padded to 7*16
#define NT     7            // number of 16-wide N tiles
#define EPS    1e-5f
#define SLOPE  0.01f

#define WAVES       8       // waves per block
#define TPW         4       // batch tiles per wave
#define TPB         (WAVES * TPW)          // 32 tiles / block
#define BLK_PER_E   (B_SZ / 16 / TPB)      // 8 blocks per edge

#define S1  72              // LDS stride (bf16) for Wt1 rows: 144B, 16B aligned
#define S2  136             // LDS stride (bf16) for Wt2 rows: 272B, 16B aligned

typedef __attribute__((ext_vector_type(16))) __bf16 v16bf;
typedef __attribute__((ext_vector_type(8)))  __bf16 v8bf;
typedef __attribute__((ext_vector_type(8)))  float  v8f;
typedef __attribute__((ext_vector_type(4)))  float  v4f;
typedef __attribute__((ext_vector_type(4)))  unsigned int v4u;
typedef __attribute__((ext_vector_type(8)))  int    v8i;
typedef __attribute__((ext_vector_type(4)))  int    v4i;

union BF16x16 { v16bf v; v8bf h[2]; };

// compiler-only fence; LDS ops from one wave complete in order (DScnt in-order)
#define WAVE_LDS_FENCE() asm volatile("" ::: "memory")

#if defined(__has_builtin)
#if __has_builtin(__builtin_amdgcn_tensor_load_to_lds) && \
    __has_builtin(__builtin_amdgcn_s_wait_tensorcnt)
#define HAVE_TDM 1
#endif
#endif

#ifdef HAVE_TDM
// Issue one TDM DMA: 2-D tile of 16 rows x 64 fp32 from x into LDS.
// D# packing per CDNA5 ISA.
// 6-arg builtin (clang-23 / therock-10.0 headers): (g0, g1, g2, g3, g4, cpol)
__device__ __forceinline__ void tdm_load_x_tile(const float* gsrc,
                                                unsigned int lds_off) {
    const uint64_t ga = (uint64_t)(uintptr_t)gsrc;
    v4u g0;
    g0[0] = 1u;                                   // count=1, user D#
    g0[1] = lds_off;                              // lds_addr (bytes)
    g0[2] = (unsigned int)ga;                     // global_addr[31:0]
    g0[3] = (unsigned int)(ga >> 32) | (2u << 30);// global_addr[56:32] | type=2
    v8i g1;
    g1[0] = 0x00020000;                           // data_size=2 (4B), no opts
    g1[1] = (int)(64u << 16);                     // tensor_dim0=64 (bits 63:48)
    g1[2] = (int)(16u << 16);                     // tensor_dim1=16 (bits 95:80)
    g1[3] = (int)(64u << 16);                     // tile_dim0=64  (bits 127:112)
    g1[4] = 16;                                   // tile_dim1=16  (bits 143:128)
    g1[5] = E_SZ * IP_SZ;                         // dim0_stride=8192 (bits 191:160)
    g1[6] = 0;
    g1[7] = 0;
    const v4i z4 = {0, 0, 0, 0};                  // group 2/3 unused (2-D tile)
    const v8i z8 = {0, 0, 0, 0, 0, 0, 0, 0};
    __builtin_amdgcn_tensor_load_to_lds(g0, g1, z4, z4, z8, 0);
}
#endif

__global__ __launch_bounds__(WAVES * 32) void edge_mlp_kernel(
    const float* __restrict__ x,  const float* __restrict__ W1,
    const float* __restrict__ b1, const float* __restrict__ g1,
    const float* __restrict__ be1,const float* __restrict__ W2,
    const float* __restrict__ b2, const float* __restrict__ g2,
    const float* __restrict__ be2,const float* __restrict__ Wo,
    const float* __restrict__ bo, float* __restrict__ out)
{
    // ---- LDS: shared weight stage (bf16, transposed, zero-padded) ----------
    __shared__ __bf16 st1[HP * S1];          // Wt1[n][k], n<112, k<64   (16.1 KB)
    __shared__ __bf16 st2[HP * S2];          // Wt2[n][k], n<112, k<128  (30.5 KB)
    __shared__ float  sb1[HP], sg1[HP], sbe1[HP];
    __shared__ float  sb2[HP], sg2[HP], sbe2[HP], swo[HP];
    // ---- LDS: per-wave h buffer + LN stats ---------------------------------
    __shared__ float  sh[WAVES][16 * HP];    // 57.3 KB
    __shared__ float  smean[WAVES][16];
    __shared__ float  srstd[WAVES][16];
#ifdef HAVE_TDM
    __shared__ float  xbuf[WAVES][16 * IP_SZ]; // 32 KB: per-wave x staging
#endif

    const int tid  = threadIdx.x;
    const int w    = tid >> 5;
    const int lane = tid & 31;
    const int half = lane >> 4;
    const int lr   = lane & 15;

    const int e  = blockIdx.x & (E_SZ - 1);
    const int tb = blockIdx.x >> 7;          // 0..BLK_PER_E-1

    const float* W1e = W1 + (size_t)e * IP_SZ * H_SZ;
    const float* W2e = W2 + (size_t)e * H_SZ * H_SZ;
    const int tile0 = tb * TPB + w * TPW;    // first batch tile of this wave

#ifdef HAVE_TDM
    const unsigned int xbuf_off = (unsigned int)(uintptr_t)(void*)&xbuf[w][0];
    // Prefetch tile 0 via Tensor Data Mover; overlaps with weight staging.
    tdm_load_x_tile(x + ((size_t)(tile0 * 16) * E_SZ + e) * IP_SZ, xbuf_off);
#endif

    // ---------------- cooperative staging (coalesced in n) ------------------
    for (int i = tid; i < HP * IP_SZ; i += WAVES * 32) {
        const int n = i % HP, k = i / HP;
        const float v = (n < H_SZ) ? W1e[k * H_SZ + n] : 0.0f;
        st1[n * S1 + k] = (__bf16)v;
    }
    for (int i = tid; i < HP * 128; i += WAVES * 32) {
        const int n = i % HP, k = i / HP;
        const float v = (n < H_SZ && k < H_SZ) ? W2e[k * H_SZ + n] : 0.0f;
        st2[n * S2 + k] = (__bf16)v;
    }
    if (tid < HP) {
        const bool ok = tid < H_SZ;
        const int  o  = e * H_SZ + tid;
        sb1[tid]  = ok ? b1[o]  : 0.0f;
        sg1[tid]  = ok ? g1[o]  : 0.0f;
        sbe1[tid] = ok ? be1[o] : 0.0f;
        sb2[tid]  = ok ? b2[o]  : 0.0f;
        sg2[tid]  = ok ? g2[o]  : 0.0f;
        sbe2[tid] = ok ? be2[o] : 0.0f;
        swo[tid]  = ok ? Wo[o]  : 0.0f;
    }
    __syncthreads();

    float* shw = &sh[w][0];
    const float bo_e = bo[e];

#pragma unroll 1
    for (int t = 0; t < TPW; ++t) {
        const int m0 = (tile0 + t) * 16;

        // ---------- A1: x tile [16 x 64] as 2 bf16 K-chunks ------------------
        // A layout: lane=row(lr); elem j -> K = c*32 + half*8 + (j<8 ? j : j+8)
        v16bf a1[2];
        {
#ifdef HAVE_TDM
            __builtin_amdgcn_s_wait_tensorcnt(0);   // tile t resident in xbuf
            WAVE_LDS_FENCE();
            const float* xrow = &xbuf[w][lr * IP_SZ];
#else
            const float* xrow = x + ((size_t)(m0 + lr) * E_SZ + e) * IP_SZ;
#endif
#pragma unroll
            for (int c = 0; c < 2; ++c) {
                const v4f p0 = *(const v4f*)(xrow + c * 32 + half * 8);
                const v4f p1 = *(const v4f*)(xrow + c * 32 + half * 8 + 4);
                const v4f p2 = *(const v4f*)(xrow + c * 32 + 16 + half * 8);
                const v4f p3 = *(const v4f*)(xrow + c * 32 + 16 + half * 8 + 4);
#pragma unroll
                for (int j = 0; j < 4; ++j) {
                    a1[c][j]      = (__bf16)p0[j];
                    a1[c][j + 4]  = (__bf16)p1[j];
                    a1[c][j + 8]  = (__bf16)p2[j];
                    a1[c][j + 12] = (__bf16)p3[j];
                }
            }
#ifdef HAVE_TDM
            // a1 is in registers; make sure the ds reads of xbuf retired,
            // then DMA the next tile over it (overlaps with GEMM1/LN/GEMM2).
            if (t + 1 < TPW) {
                asm volatile("s_wait_dscnt 0x0" ::: "memory");
                tdm_load_x_tile(
                    x + ((size_t)((tile0 + t + 1) * 16) * E_SZ + e) * IP_SZ,
                    xbuf_off);
            }
#endif
        }

        // ---------- GEMM1 + bias + LeakyReLU -> sh ---------------------------
        // B layout: lane=col(lr); elem j -> K = c*32 + half*8 + (j<8 ? j : j+8)
#pragma unroll
        for (int nt = 0; nt < NT; ++nt) {
            const int ncol = nt * 16 + lr;
            const __bf16* bp0 = &st1[ncol * S1 + half * 8];        // chunk 0
            const __bf16* bp1 = &st1[ncol * S1 + 32 + half * 8];   // chunk 1
            BF16x16 bf0, bf1;
            bf0.h[0] = *(const v8bf*)(bp0);
            bf0.h[1] = *(const v8bf*)(bp0 + 16);
            bf1.h[0] = *(const v8bf*)(bp1);
            bf1.h[1] = *(const v8bf*)(bp1 + 16);
            v8f acc = {};
            acc = __builtin_amdgcn_wmma_f32_16x16x32_bf16(
                      false, a1[0], false, bf0.v, (short)0, acc, false, false);
            acc = __builtin_amdgcn_wmma_f32_16x16x32_bf16(
                      false, a1[1], false, bf1.v, (short)0, acc, false, false);
            const float bias = sb1[ncol];
#pragma unroll
            for (int r = 0; r < 8; ++r) {
                float v = acc[r] + bias;
                v = (v > 0.0f) ? v : SLOPE * v;       // padded cols stay ~0
                shw[(r + half * 8) * HP + ncol] = v;
            }
        }
        WAVE_LDS_FENCE();

        // ---------- LN1 stats (biased variance) ------------------------------
        {
            float s = 0.0f, sq = 0.0f;
#pragma unroll
            for (int k4 = 0; k4 < H_SZ / 4; ++k4) {
                const v4f v = *(const v4f*)&shw[lr * HP + k4 * 4];
#pragma unroll
                for (int j = 0; j < 4; ++j) { s += v[j]; sq += v[j] * v[j]; }
            }
            const float m   = s * (1.0f / H_SZ);
            const float var = sq * (1.0f / H_SZ) - m * m;
            const float rs  = rsqrtf(var + EPS);
            if (lane < 16) { smean[w][lr] = m; srstd[w][lr] = rs; }
        }
        WAVE_LDS_FENCE();

        // ---------- A2: LN(h1) as 4 bf16 K-chunks ----------------------------
        v16bf a2[4];
        {
            const float m  = smean[w][lr];
            const float rs = srstd[w][lr];
#pragma unroll
            for (int c = 0; c < 4; ++c) {
#pragma unroll
                for (int g = 0; g < 2; ++g) {         // two contiguous groups of 8
                    const int k0 = c * 32 + half * 8 + g * 16;
                    const v4f hv0 = *(const v4f*)&shw[lr * HP + k0];
                    const v4f hv1 = *(const v4f*)&shw[lr * HP + k0 + 4];
                    const v4f gg0 = *(const v4f*)&sg1[k0];
                    const v4f gg1 = *(const v4f*)&sg1[k0 + 4];
                    const v4f bb0 = *(const v4f*)&sbe1[k0];
                    const v4f bb1 = *(const v4f*)&sbe1[k0 + 4];
#pragma unroll
                    for (int j = 0; j < 4; ++j) {
                        a2[c][g * 8 + j]     = (__bf16)((hv0[j] - m) * rs * gg0[j] + bb0[j]);
                        a2[c][g * 8 + j + 4] = (__bf16)((hv1[j] - m) * rs * gg1[j] + bb1[j]);
                    }
                }
            }
        }
        WAVE_LDS_FENCE();   // done reading sh before GEMM2 overwrites it

        // ---------- GEMM2 + bias + LeakyReLU -> sh ---------------------------
#pragma unroll
        for (int nt = 0; nt < NT; ++nt) {
            const int ncol = nt * 16 + lr;
            v8f acc = {};
#pragma unroll
            for (int c = 0; c < 4; ++c) {
                const __bf16* bp = &st2[ncol * S2 + c * 32 + half * 8];
                BF16x16 bf;
                bf.h[0] = *(const v8bf*)(bp);
                bf.h[1] = *(const v8bf*)(bp + 16);
                acc = __builtin_amdgcn_wmma_f32_16x16x32_bf16(
                          false, a2[c], false, bf.v, (short)0, acc, false, false);
            }
            const float bias = sb2[ncol];
#pragma unroll
            for (int r = 0; r < 8; ++r) {
                float v = acc[r] + bias;
                v = (v > 0.0f) ? v : SLOPE * v;
                shw[(r + half * 8) * HP + ncol] = v;
            }
        }
        WAVE_LDS_FENCE();

        // ---------- LN2 stats + head dot -------------------------------------
        {
            float s = 0.0f, sq = 0.0f;
#pragma unroll
            for (int k4 = 0; k4 < H_SZ / 4; ++k4) {
                const v4f v = *(const v4f*)&shw[lr * HP + k4 * 4];
#pragma unroll
                for (int j = 0; j < 4; ++j) { s += v[j]; sq += v[j] * v[j]; }
            }
            const float m   = s * (1.0f / H_SZ);
            const float var = sq * (1.0f / H_SZ) - m * m;
            const float rs  = rsqrtf(var + EPS);

            if (lane < 16) {
                float acc = 0.0f;
#pragma unroll
                for (int k4 = 0; k4 < H_SZ / 4; ++k4) {
                    const v4f hv = *(const v4f*)&shw[lr * HP + k4 * 4];
                    const v4f gg = *(const v4f*)&sg2[k4 * 4];
                    const v4f bb = *(const v4f*)&sbe2[k4 * 4];
                    const v4f wo = *(const v4f*)&swo[k4 * 4];
#pragma unroll
                    for (int j = 0; j < 4; ++j)
                        acc += ((hv[j] - m) * rs * gg[j] + bb[j]) * wo[j];
                }
                // reference scramble: transpose(B,E)->reshape => edge-major flat
                out[(size_t)e * B_SZ + (m0 + lr)] = acc + bo_e;
            }
        }
        WAVE_LDS_FENCE();
    }
}

extern "C" void kernel_launch(void* const* d_in, const int* in_sizes, int n_in,
                              void* d_out, int out_size, void* d_ws, size_t ws_size,
                              hipStream_t stream) {
    const float* x   = (const float*)d_in[0];
    const float* W1  = (const float*)d_in[1];
    const float* b1  = (const float*)d_in[2];
    const float* g1  = (const float*)d_in[3];
    const float* be1 = (const float*)d_in[4];
    const float* W2  = (const float*)d_in[5];
    const float* b2  = (const float*)d_in[6];
    const float* g2  = (const float*)d_in[7];
    const float* be2 = (const float*)d_in[8];
    const float* Wo  = (const float*)d_in[9];
    const float* bo  = (const float*)d_in[10];
    float* out = (float*)d_out;

    dim3 grid(E_SZ * BLK_PER_E);     // 1024 blocks
    dim3 block(WAVES * 32);          // 8 wave32 per block
    edge_mlp_kernel<<<grid, block, 0, stream>>>(
        x, W1, b1, g1, be1, W2, b2, g2, be2, Wo, bo, out);
}
